// SSNTransformerDecoderLayer_9603546873994
// MI455X (gfx1250) — matrix operand order repaired and checked by reading, not compiled, not run on hardware
//
#include <hip/hip_runtime.h>
#include <hip/hip_bf16.h>

// Problem constants (fixed by the reference: T=1024, B=8, D=1024, H=8)
#define TT 1024
#define BB 8
#define DD 1024
#define HH 8
#define HD 128
#define BHN 64   // B*H

typedef __attribute__((ext_vector_type(16))) __bf16 v16bf;
typedef __attribute__((ext_vector_type(8)))  float  v8f;
typedef __attribute__((ext_vector_type(4)))  float  f4;
typedef __attribute__((ext_vector_type(4)))  unsigned int u4;

union FragU { v16bf v; u4 q[2]; };

__device__ __forceinline__ __bf16 f2bf(float f) {
  unsigned u = __builtin_bit_cast(unsigned, f);
  u += 0x7FFFu + ((u >> 16) & 1u);            // round-to-nearest-even
  unsigned short h = (unsigned short)(u >> 16);
  return __builtin_bit_cast(__bf16, h);
}

__device__ __forceinline__ v8f wmma_bf16(v16bf a, v16bf b, v8f c) {
  // D = A(16x32 bf16) * B(32x16 bf16) + C(16x16 f32)
  return __builtin_amdgcn_wmma_f32_16x16x32_bf16(false, a, false, b, (short)0, c,
                                                 false, false);
}

// ---- fragment loaders (bf16 only in hot loops) ------------------------------
// A layout (16-bit, 16x32): lane L (m = L&15, half = L>>4):
//   elems 0..7  <-> K = half*8 + 0..7 ; elems 8..15 <-> K = 16 + half*8 + 0..7
// B layout (16-bit, 32x16): lane L (n = L&15, half = L>>4):
//   elems 0..7  <-> K = half*8 + 0..7 ; elems 8..15 <-> K = 16 + half*8 + 0..7

__device__ __forceinline__ v16bf load_a_bf(const __bf16* row, int kc, int half) {
  FragU u;
  u.q[0] = *(const u4*)(row + kc + half * 8);
  u.q[1] = *(const u4*)(row + kc + 16 + half * 8);
  return u.v;
}

__device__ __forceinline__ v16bf load_b_bf(const __bf16* row, int kc, int half) {
  FragU u;
  const __bf16* p = row + kc + half * 8;
  u.q[0] = *(const u4*)(p);
  u.q[1] = *(const u4*)(p + 16);
  return u.v;
}

// ---- elementwise f32 -> bf16 (8 elements/thread, vector load + b128 store) --
__global__ __launch_bounds__(256) void cvt_bf_kernel(const float* __restrict__ src,
                                                     __bf16* __restrict__ dst,
                                                     int n8) {
  const int i = blockIdx.x * 256 + threadIdx.x;
  if (i >= n8) return;
  const f4 a = *(const f4*)(src + (long)i * 8);
  const f4 b = *(const f4*)(src + (long)i * 8 + 4);
  union { __bf16 h[8]; u4 q; } u;
  u.h[0] = f2bf(a.x); u.h[1] = f2bf(a.y); u.h[2] = f2bf(a.z); u.h[3] = f2bf(a.w);
  u.h[4] = f2bf(b.x); u.h[5] = f2bf(b.y); u.h[6] = f2bf(b.z); u.h[7] = f2bf(b.w);
  *(u4*)(dst + (long)i * 8) = u.q;
}

// ---- prep: fold biaffine into the query projection (bf16 output) ------------
// wq2[jg][k] = sum_i Wbi[i,0,j] * Wq[h*128+i][k]   (jg = h*128+j)
__global__ __launch_bounds__(256) void prep_w_kernel(const float* __restrict__ Wq,
                                                     const float* __restrict__ Wbi,
                                                     __bf16* __restrict__ wq2) {
  const int idx = blockIdx.x * 256 + threadIdx.x;   // 1M threads
  const int k = idx & 1023;
  const int jg = idx >> 10;
  const int h = jg >> 7, j = jg & 127;
  float s = 0.f;
  for (int i = 0; i < 128; ++i)
    s += Wbi[i * 129 + j] * Wq[(h * 128 + i) * DD + k];
  wq2[jg * DD + k] = f2bf(s);
}

// bias2[jg] = Wbi[128,0,j] + sum_i bq[h*128+i] * Wbi[i,0,j]
__global__ __launch_bounds__(256) void prep_b_kernel(const float* __restrict__ bq,
                                                     const float* __restrict__ Wbi,
                                                     float* __restrict__ bias2) {
  const int jg = blockIdx.x * 256 + threadIdx.x;    // 1024 threads
  const int h = jg >> 7, j = jg & 127;
  float s = Wbi[128 * 129 + j];
  for (int i = 0; i < 128; ++i) s += bq[h * 128 + i] * Wbi[i * 129 + j];
  bias2[jg] = s;
}

// ---- projections: q2 / K / V^T, all as (tgt @ W^T + b), bf16 in, bf16 out ---
// mode 0: q2 (head-major [bh][t][hd]);  mode 1: K same;  mode 2: V transposed [bh][hd][t]
__global__ __launch_bounds__(256) void qkv_kernel(
    const __bf16* __restrict__ tgtb,
    const __bf16* __restrict__ wq2, const float* __restrict__ bias2,
    const __bf16* __restrict__ wkb, const float* __restrict__ bk,
    const __bf16* __restrict__ wvb, const float* __restrict__ bv,
    __bf16* __restrict__ q2, __bf16* __restrict__ kdst, __bf16* __restrict__ vtd) {
  const int mode = blockIdx.z;
  const __bf16* W  = (mode == 0) ? wq2 : (mode == 1) ? wkb : wvb;
  const float*  bi = (mode == 0) ? bias2 : (mode == 1) ? bk : bv;
  const int lane = threadIdx.x & 31, wave = threadIdx.x >> 5;
  const int half = lane >> 4, l15 = lane & 15;
  const int mbase = blockIdx.x * 128 + wave * 16;   // row tile in (T*B)
  const int nbase = blockIdx.y * 128;               // col tile in D
  const __bf16* arow = tgtb + (long)(mbase + l15) * DD;

  const __bf16* brow[8];
#pragma unroll
  for (int nt = 0; nt < 8; ++nt)
    brow[nt] = W + (long)(nbase + nt * 16 + l15) * DD;

  v8f c[8] = {};
  for (int kc = 0; kc < DD; kc += 32) {             // pure b128 loads + wmma
    v16bf a = load_a_bf(arow, kc, half);
#pragma unroll
    for (int nt = 0; nt < 8; ++nt)
      c[nt] = wmma_bf16(a, load_b_bf(brow[nt], kc, half), c[nt]);
  }
#pragma unroll
  for (int nt = 0; nt < 8; ++nt) {
    const int col = nbase + nt * 16 + l15;
    const float bb = bi[col];
    const int h = col >> 7, dl = col & 127;
#pragma unroll
    for (int r = 0; r < 8; ++r) {
      const int m = mbase + r + 8 * half;           // m = t*B + b
      const int t = m >> 3, b = m & 7;
      const __bf16 x = f2bf(c[nt][r] + bb);
      if (mode == 2)
        vtd[((long)(b * HH + h) * HD + dl) * TT + t] = x;
      else if (mode == 1)
        kdst[((long)(b * HH + h) * TT + t) * HD + dl] = x;
      else
        q2[((long)(b * HH + h) * TT + t) * HD + dl] = x;
    }
  }
}

// ---- flash attention: scores = q2 . k, online softmax, O = P @ V ------------
__global__ __launch_bounds__(256) void attn_kernel(
    const __bf16* __restrict__ q2, const __bf16* __restrict__ kbf,
    const __bf16* __restrict__ vt, const unsigned char* __restrict__ mask,
    float* __restrict__ out) {
  __shared__ alignas(16) __bf16 ldsP[8][16][136];   // per-wave-private P staging

  const int lane = threadIdx.x & 31, wave = threadIdx.x >> 5;
  const int half = lane >> 4, l15 = lane & 15;
  const int bh = blockIdx.y;
  const int b = bh >> 3, h = bh & 7;
  const int qbase = blockIdx.x * 128 + wave * 16;

  // q2 A-fragments for this wave's 16 query rows, K=128 in 4 chunks of 32
  const __bf16* qrow = q2 + ((long)bh * TT + (qbase + l15)) * HD;
  v16bf aq[4];
#pragma unroll
  for (int kc = 0; kc < 4; ++kc) aq[kc] = load_a_bf(qrow, kc * 32, half);

  v8f o[8] = {};
  float mrun[8], lrun[8];
#pragma unroll
  for (int r = 0; r < 8; ++r) { mrun[r] = -3.0e38f; lrun[r] = 0.f; }

  for (int j = 0; j < 8; ++j) {                     // stream over keys, 128/tile
    const int jb = j * 128;

    // prefetch next K/V tiles (global_prefetch_b8): hide HBM/L2 latency under
    // this tile's WMMAs; rows are 256 B -> touch both 128-B lines per row.
    if (j < 7) {
      const int jn = jb + 128;
#pragma unroll
      for (int nt = 0; nt < 8; ++nt) {
        const __bf16* kn = kbf + ((long)bh * TT + (jn + nt * 16 + l15)) * HD;
        __builtin_prefetch(kn, 0, 3);
        __builtin_prefetch(kn + 64, 0, 3);
        const __bf16* vn = vt + ((long)bh * HD + (nt * 16 + l15)) * TT + jn;
        __builtin_prefetch(vn, 0, 3);
        __builtin_prefetch(vn + 64, 0, 3);
      }
    }

    v8f s[8];
#pragma unroll
    for (int nt = 0; nt < 8; ++nt) {
      const __bf16* krow = kbf + ((long)bh * TT + (jb + nt * 16 + l15)) * HD;
      v8f acc = {};
#pragma unroll
      for (int kc = 0; kc < 4; ++kc)
        acc = wmma_bf16(aq[kc], load_b_bf(krow, kc * 32, half), acc);
      // key-position mask (numpy bool = 1 byte); all-False in the harness setup
      if (mask[b * TT + jb + nt * 16 + l15]) {
#pragma unroll
        for (int r = 0; r < 8; ++r) acc[r] = -1.0e30f;
      }
      s[nt] = acc;
    }
    // online softmax: row m = r + 8*half lives in one VGPR across a 16-lane group
#pragma unroll
    for (int r = 0; r < 8; ++r) {
      float rm = s[0][r];
#pragma unroll
      for (int nt = 1; nt < 8; ++nt) rm = fmaxf(rm, s[nt][r]);
#pragma unroll
      for (int off = 1; off < 16; off <<= 1) rm = fmaxf(rm, __shfl_xor(rm, off, 16));
      const float mnew = fmaxf(mrun[r], rm);
      const float corr = __expf(mrun[r] - mnew);
      float ls = 0.f;
#pragma unroll
      for (int nt = 0; nt < 8; ++nt) {
        float p = __expf(s[nt][r] - mnew);
        s[nt][r] = p;
        ls += p;
      }
#pragma unroll
      for (int off = 1; off < 16; off <<= 1) ls += __shfl_xor(ls, off, 16);
      lrun[r] = lrun[r] * corr + ls;
      mrun[r] = mnew;
#pragma unroll
      for (int nt = 0; nt < 8; ++nt) o[nt][r] *= corr;
    }
    // stage P (C-layout f32 -> A-layout bf16) through per-wave LDS region
#pragma unroll
    for (int nt = 0; nt < 8; ++nt)
#pragma unroll
      for (int r = 0; r < 8; ++r)
        ldsP[wave][r + 8 * half][nt * 16 + l15] = f2bf(s[nt][r]);

    const __bf16* prow = &ldsP[wave][l15][0];
    v16bf pf[4];
#pragma unroll
    for (int kc = 0; kc < 4; ++kc) pf[kc] = load_a_bf(prow, kc * 32, half);

    // O += P @ V  (V stored transposed [bh][hd][t] -> contiguous B-fragments)
#pragma unroll
    for (int nt = 0; nt < 8; ++nt) {
      const __bf16* vrow = vt + ((long)bh * HD + (nt * 16 + l15)) * TT + jb;
#pragma unroll
      for (int kc = 0; kc < 4; ++kc)
        o[nt] = wmma_bf16(pf[kc], load_b_bf(vrow, kc * 32, half), o[nt]);
    }
  }

  float inv[8];
#pragma unroll
  for (int r = 0; r < 8; ++r) inv[r] = 1.0f / lrun[r];
#pragma unroll
  for (int nt = 0; nt < 8; ++nt) {
    const int d = h * HD + nt * 16 + l15;
#pragma unroll
    for (int r = 0; r < 8; ++r) {
      const int t = qbase + r + 8 * half;
      out[((long)t * BB + b) * DD + d] = o[nt][r] * inv[r];
    }
  }
}

extern "C" void kernel_launch(void* const* d_in, const int* in_sizes, int n_in,
                              void* d_out, int out_size, void* d_ws, size_t ws_size,
                              hipStream_t stream) {
  (void)in_sizes; (void)n_in; (void)out_size; (void)ws_size;
  const float* tgt  = (const float*)d_in[0];
  const unsigned char* mask = (const unsigned char*)d_in[1];  // numpy bool
  const float* Wq   = (const float*)d_in[2];
  const float* bq   = (const float*)d_in[3];
  const float* Wk   = (const float*)d_in[4];
  const float* bk   = (const float*)d_in[5];
  const float* Wv   = (const float*)d_in[6];
  const float* bv   = (const float*)d_in[7];
  const float* Wbi  = (const float*)d_in[8];
  // d_in[9..14] = W1,b1,W2,b2,ln_g,ln_b : FFN/LN result is discarded -> unused
  float* out = (float*)d_out;

  // workspace layout (~70 MB total, all offsets 4 KiB aligned)
  char* ws = (char*)d_ws;
  __bf16* wq2b  = (__bf16*)(ws + 0);                 //  2 MiB folded query weight
  float*  bias2 = (float*) (ws + 2097152);           //  4 KiB folded query bias
  __bf16* wkb   = (__bf16*)(ws + 2101248);           //  2 MiB Wk bf16
  __bf16* wvb   = (__bf16*)(ws + 4198400);           //  2 MiB Wv bf16
  __bf16* tgtb  = (__bf16*)(ws + 6295552);           // 16 MiB tgt bf16
  __bf16* q2    = (__bf16*)(ws + 23072768);          // 16 MiB
  __bf16* kbf   = (__bf16*)(ws + 39849984);          // 16 MiB
  __bf16* vt    = (__bf16*)(ws + 56627200);          // 16 MiB

  // one-time conversions / folds (all traffic ~40 MB, negligible vs GEMMs)
  cvt_bf_kernel<<<4096, 256, 0, stream>>>(tgt, tgtb, (TT * BB * DD) / 8);
  cvt_bf_kernel<<<512, 256, 0, stream>>>(Wk, wkb, (DD * DD) / 8);
  cvt_bf_kernel<<<512, 256, 0, stream>>>(Wv, wvb, (DD * DD) / 8);
  prep_w_kernel<<<4096, 256, 0, stream>>>(Wq, Wbi, wq2b);
  prep_b_kernel<<<4, 256, 0, stream>>>(bq, Wbi, bias2);

  qkv_kernel<<<dim3(64, 8, 3), 256, 0, stream>>>(tgtb, wq2b, bias2, wkb, bk, wvb, bv,
                                                 q2, kbf, vt);
  attn_kernel<<<dim3(8, 64), 256, 0, stream>>>(q2, kbf, vt, mask, out);
}
